// IPW_30520037605944
// MI455X (gfx1250) — matrix operand order repaired and checked by reading, not compiled, not run on hardware
//
#include <hip/hip_runtime.h>
#include <hip/hip_bf16.h>
#include <stddef.h>

#define NN   4096
#define INF  512
#define OUTF 512

typedef __attribute__((ext_vector_type(16))) __bf16       v16bf;
typedef __attribute__((ext_vector_type(8)))  float        v8f;
typedef __attribute__((ext_vector_type(4)))  float        f32x4;
typedef __attribute__((ext_vector_type(4)))  unsigned int u32x4;

union BFrag { u32x4 u[2]; v16bf bf; };

__device__ __forceinline__ unsigned short f2bf(float f) {
    unsigned u = __float_as_uint(f);
    unsigned r = u + 0x7FFFu + ((u >> 16) & 1u);   // round-to-nearest-even
    return (unsigned short)(r >> 16);
}
__device__ __forceinline__ unsigned int pack2(unsigned short lo, unsigned short hi) {
    return (unsigned int)lo | ((unsigned int)hi << 16);
}

#if __has_builtin(__builtin_amdgcn_cvt_pk_bf16_f32)
__device__ __forceinline__ unsigned int cvtpk(float a, float b) {
    typedef __attribute__((ext_vector_type(2))) __bf16 v2bf;
    union { v2bf v; unsigned int u; } c;
    c.v = __builtin_amdgcn_cvt_pk_bf16_f32(a, b);
    return c.u;
}
#else
__device__ __forceinline__ unsigned int cvtpk(float a, float b) {
    return pack2(f2bf(a), f2bf(b));
}
#endif

// Async global->LDS copy of 16B (per lane). INST_OFFSET advances both sides.
__device__ __forceinline__ void async_b128(unsigned lds_off, unsigned long long gaddr) {
    asm volatile("global_load_async_to_lds_b128 %0, %1, off" :: "v"(lds_off), "v"(gaddr) : "memory");
}
__device__ __forceinline__ void wait_async0() {
    asm volatile("s_wait_asynccnt 0x0" ::: "memory");
}

// ---------------------------------------------------------------------------
// Kernel 1: supT[c][j] = (X @ W)[j][c] in bf16   (M=4096, N=512, K=512)
// Block tile 64(M) x 256(N), K-step 32. 8 waves (2x4), 32x64 per wave.
// ---------------------------------------------------------------------------
__global__ __launch_bounds__(256) void k_support_T(
    const float* __restrict__ X, const float* __restrict__ W,
    unsigned short* __restrict__ supT)
{
    __shared__ __align__(16) unsigned short lds_a[64 * 40];   // [row][k], stride 40
    __shared__ __align__(16) unsigned short lds_b[256 * 40];  // [col][k], stride 40

    const int t    = threadIdx.x;
    const int lane = t & 31, wid = t >> 5;
    const int wm   = wid >> 2, wn = wid & 3;
    const int half = lane >> 4, l16 = lane & 15;
    const int i0   = blockIdx.x * 64;
    const int c0   = blockIdx.y * 256;
    const int ar   = t >> 2,  ak0 = (t & 3) * 8;
    const int bk   = t >> 3,  bc0 = (t & 7) * 32;

    v8f acc[2][4] = {};

    for (int kk = 0; kk < INF; kk += 32) {
        { // A tile: X[i0+ar][kk+ak0 .. +7] -> bf16
            const float* xp = X + (size_t)(i0 + ar) * INF + kk + ak0;
            f32x4 x0 = *(const f32x4*)xp;
            f32x4 x1 = *(const f32x4*)(xp + 4);
            u32x4 v;
            v.x = cvtpk(x0.x, x0.y); v.y = cvtpk(x0.z, x0.w);
            v.z = cvtpk(x1.x, x1.y); v.w = cvtpk(x1.z, x1.w);
            *(u32x4*)&lds_a[ar * 40 + ak0] = v;
        }
        { // B tile: W[kk+bk][c0+bc0 .. +31] -> lds_b[c][k] (transpose)
            const float* wp = W + (size_t)(kk + bk) * OUTF + c0 + bc0;
            #pragma unroll
            for (int q4 = 0; q4 < 8; ++q4) {
                f32x4 v = ((const f32x4*)wp)[q4];
                lds_b[(bc0 + q4 * 4 + 0) * 40 + bk] = f2bf(v.x);
                lds_b[(bc0 + q4 * 4 + 1) * 40 + bk] = f2bf(v.y);
                lds_b[(bc0 + q4 * 4 + 2) * 40 + bk] = f2bf(v.z);
                lds_b[(bc0 + q4 * 4 + 3) * 40 + bk] = f2bf(v.w);
            }
        }
        __syncthreads();

        BFrag afrag[2], bfrag[4];
        #pragma unroll
        for (int tm = 0; tm < 2; ++tm) {   // A: lanes<16 K{0-7,16-23}; lanes>=16 K{8-15,24-31}
            const u32x4* p = (const u32x4*)&lds_a[(wm * 32 + tm * 16 + l16) * 40];
            afrag[tm].u[0] = p[half];
            afrag[tm].u[1] = p[2 + half];
        }
        #pragma unroll
        for (int tn = 0; tn < 4; ++tn) {   // B: lanes<16 K{0-7,16-23}; lanes>=16 K{8-15,24-31}
            const u32x4* p = (const u32x4*)&lds_b[(wn * 64 + tn * 16 + l16) * 40];
            bfrag[tn].u[0] = p[half];
            bfrag[tn].u[1] = p[2 + half];
        }
        #pragma unroll
        for (int tm = 0; tm < 2; ++tm)
            #pragma unroll
            for (int tn = 0; tn < 4; ++tn)
                acc[tm][tn] = __builtin_amdgcn_wmma_f32_16x16x32_bf16(
                    false, afrag[tm].bf, false, bfrag[tn].bf,
                    (short)0, acc[tm][tn], false, false);
        __syncthreads();
    }

    // Epilogue: write transposed bf16 (consecutive C-regs are consecutive j -> 16B stores)
    #pragma unroll
    for (int tm = 0; tm < 2; ++tm)
        #pragma unroll
        for (int tn = 0; tn < 4; ++tn) {
            const int cG = c0 + wn * 64 + tn * 16 + l16;
            const int jb = i0 + wm * 32 + tm * 16 + half * 8;
            u32x4 v;
            v.x = cvtpk(acc[tm][tn][0], acc[tm][tn][1]);
            v.y = cvtpk(acc[tm][tn][2], acc[tm][tn][3]);
            v.z = cvtpk(acc[tm][tn][4], acc[tm][tn][5]);
            v.w = cvtpk(acc[tm][tn][6], acc[tm][tn][7]);
            *(u32x4*)(supT + (size_t)cG * NN + jb) = v;
        }
}

// ---------------------------------------------------------------------------
// Kernel 2: out[i][c] = sum_j proj(E[i][j]) * support[j][c] + bias[c]
// Fused edge projection; double-buffered LDS; async global->LDS for B tiles.
// Block tile 64(M) x 256(N), K-step 32, 128 iterations.
// ---------------------------------------------------------------------------
__global__ __launch_bounds__(256) void k_propagate(
    const float* __restrict__ E, const float* __restrict__ wq,
    const unsigned short* __restrict__ supT, const float* __restrict__ bias,
    float* __restrict__ out)
{
    __shared__ __align__(16) unsigned short lds_a[2][64 * 40];
    __shared__ __align__(16) unsigned short lds_b[2][256 * 40];

    const int t    = threadIdx.x;
    const int lane = t & 31, wid = t >> 5;
    const int wm   = wid >> 2, wn = wid & 3;
    const int half = lane >> 4, l16 = lane & 15;
    const int i0   = blockIdx.x * 64;
    const int c0   = blockIdx.y * 256;
    const int ar   = t >> 2, ak0 = (t & 3) * 8;

    const float w0 = wq[0], w1 = wq[1], w2 = wq[2], w3 = wq[3];

    v8f acc[2][4] = {};
    f32x4 ebuf[8];

    // --- helpers -----------------------------------------------------------
    auto loadE = [&](int j0) {
        const f32x4* ep = (const f32x4*)E + ((size_t)(i0 + ar) * NN + j0 + ak0);
        #pragma unroll
        for (int q = 0; q < 8; ++q) ebuf[q] = ep[q];
    };
    auto projStore = [&](int buf) {
        float a[8];
        #pragma unroll
        for (int q = 0; q < 8; ++q)
            a[q] = ebuf[q].x * w0 + ebuf[q].y * w1 + ebuf[q].z * w2 + ebuf[q].w * w3;
        u32x4 v;
        v.x = cvtpk(a[0], a[1]); v.y = cvtpk(a[2], a[3]);
        v.z = cvtpk(a[4], a[5]); v.w = cvtpk(a[6], a[7]);
        *(u32x4*)&lds_a[buf][ar * 40 + ak0] = v;
    };
    auto asyncB = [&](int j0, int buf) {
        unsigned long long ga = (unsigned long long)(size_t)(supT + (size_t)(c0 + t) * NN + j0);
        unsigned lo = (unsigned)(size_t)&lds_b[buf][t * 40];  // LDS aperture low bits are 0
        asm volatile(
            "global_load_async_to_lds_b128 %0, %1, off\n\t"
            "global_load_async_to_lds_b128 %0, %1, off offset:16\n\t"
            "global_load_async_to_lds_b128 %0, %1, off offset:32\n\t"
            "global_load_async_to_lds_b128 %0, %1, off offset:48"
            :: "v"(lo), "v"(ga) : "memory");
    };
    // -----------------------------------------------------------------------

    // prologue: stage tile 0 into buffer 0
    loadE(0);
    asyncB(0, 0);
    projStore(0);
    wait_async0();
    __syncthreads();

    for (int it = 0; it < NN / 32; ++it) {
        const int cur = it & 1;
        const int j1  = (it + 1) * 32;
        if (j1 < NN) {           // issue next tile's global traffic early
            loadE(j1);
            asyncB(j1, cur ^ 1);
        }

        BFrag afrag[2], bfrag[4];
        #pragma unroll
        for (int tm = 0; tm < 2; ++tm) {
            const u32x4* p = (const u32x4*)&lds_a[cur][(wm * 32 + tm * 16 + l16) * 40];
            afrag[tm].u[0] = p[half];
            afrag[tm].u[1] = p[2 + half];
        }
        #pragma unroll
        for (int tn = 0; tn < 4; ++tn) {
            const u32x4* p = (const u32x4*)&lds_b[cur][(wn * 64 + tn * 16 + l16) * 40];
            bfrag[tn].u[0] = p[half];
            bfrag[tn].u[1] = p[2 + half];
        }
        #pragma unroll
        for (int tm = 0; tm < 2; ++tm)
            #pragma unroll
            for (int tn = 0; tn < 4; ++tn)
                acc[tm][tn] = __builtin_amdgcn_wmma_f32_16x16x32_bf16(
                    false, afrag[tm].bf, false, bfrag[tn].bf,
                    (short)0, acc[tm][tn], false, false);

        if (j1 < NN) projStore(cur ^ 1);   // after WMMAs: loads had time to land
        wait_async0();
        __syncthreads();
    }

    // Epilogue: bias add, f32 stores (16 lanes -> 64B coalesced per row-half)
    #pragma unroll
    for (int tn = 0; tn < 4; ++tn) {
        const int col = c0 + wn * 64 + tn * 16 + l16;
        const float b = bias[col];
        #pragma unroll
        for (int tm = 0; tm < 2; ++tm) {
            const size_t base = (size_t)(i0 + wm * 32 + tm * 16 + half * 8) * OUTF + col;
            #pragma unroll
            for (int r = 0; r < 8; ++r)
                out[base + (size_t)r * OUTF] = acc[tm][tn][r] + b;
        }
    }
}

extern "C" void kernel_launch(void* const* d_in, const int* in_sizes, int n_in,
                              void* d_out, int out_size, void* d_ws, size_t ws_size,
                              hipStream_t stream) {
    (void)in_sizes; (void)n_in; (void)out_size; (void)ws_size;
    const float* X    = (const float*)d_in[0];
    // d_in[1] = adj (unused by forward)
    const float* E    = (const float*)d_in[2];
    const float* W    = (const float*)d_in[3];
    const float* wq   = (const float*)d_in[4];
    const float* bias = (const float*)d_in[5];
    float*          out  = (float*)d_out;
    unsigned short* supT = (unsigned short*)d_ws;   // 512*4096 bf16 = 4 MB

    k_support_T<<<dim3(NN / 64, OUTF / 256), 256, 0, stream>>>(X, W, supT);
    k_propagate<<<dim3(NN / 64, OUTF / 256), 256, 0, stream>>>(E, wq, supT, bias, out);
}
